// CausalMultiHeadSelfAttention_67095979098214
// MI455X (gfx1250) — matrix operand order repaired and checked by reading, not compiled, not run on hardware
//
#include <hip/hip_runtime.h>
#include <hip/hip_bf16.h>

// ---------------------------------------------------------------------------
// Causal MHA forward for MI455X (gfx1250), wave32 + v_wmma_f32_16x16x32_f16.
// Compute-bound (~0.4 TFLOP vs ~hundreds of MB of traffic @23.3TB/s), so all
// GEMMs (QKV proj, QK^T, P*V, out proj) run on the WMMA pipe with f16 inputs
// and f32 accumulation.
// ---------------------------------------------------------------------------

#define Bdim 4
#define Sdim 2048
#define Ddim 2048
#define Hn   16
#define DKd  128          // D / H
#define Mrows (Bdim * Sdim)   // 8192

typedef _Float16 h16;
typedef __attribute__((ext_vector_type(16))) _Float16 v16h;
typedef __attribute__((ext_vector_type(8)))  _Float16 v8h;
typedef __attribute__((ext_vector_type(8)))  float    v8f;

union AFrag { v16h v; v8h h[2]; };

__device__ __forceinline__ v8f v8f_splat(float x) {
  v8f r;
  #pragma unroll
  for (int i = 0; i < 8; ++i) r[i] = x;
  return r;
}

// A-matrix fragment (16x32 f16, M=row, K contiguous in memory, leading dim ld).
// ISA layout: lanes 0-15: K[0:8] then K[16:24]; lanes 16-31: K[8:16], K[24:32].
__device__ __forceinline__ v16h load_a_frag(const h16* base, int ld, int lane) {
  int r  = lane & 15;
  int hh = lane >> 4;
  const h16* row = base + (size_t)r * ld;
  AFrag f;
  f.h[0] = *(const v8h*)(row + hh * 8);
  f.h[1] = *(const v8h*)(row + 16 + hh * 8);
  return f.v;
}

// B-matrix fragment (32x16 f16). Source stored as B^T row-major: [n, k],
// K contiguous. ISA layout: lanes 0-15: K[0:8] then K[16:24]; lanes 16-31: K[8:16], K[24:32].
__device__ __forceinline__ v16h load_b_frag(const h16* base, int ld, int lane) {
  int n  = lane & 15;
  int hh = lane >> 4;
  const h16* row = base + (size_t)n * ld;
  AFrag f;
  f.h[0] = *(const v8h*)(row + hh * 8);
  f.h[1] = *(const v8h*)(row + 16 + hh * 8);
  return f.v;
}

// ---------------------------------------------------------------------------
// fp32 -> f16 conversion
// ---------------------------------------------------------------------------
__global__ void cvt_f32_to_f16(const float* __restrict__ in,
                               h16* __restrict__ out, int n) {
  int i = blockIdx.x * blockDim.x + threadIdx.x;
  if (i < n) out[i] = (h16)in[i];
}

// ---------------------------------------------------------------------------
// GEMM: C[MxN] = A[MxK] * W^T, W stored [N,K] row-major (einsum bsi,oi->bso).
// Block = 4 waves (128 thr); each wave computes a 32x64 tile; block = 64x128.
// ---------------------------------------------------------------------------
template <bool F16OUT>
__global__ __launch_bounds__(128) void gemm_xwt_kernel(
    const h16* __restrict__ A, const h16* __restrict__ W,
    void* __restrict__ Cout, int M, int N, int K) {
  int lane = threadIdx.x & 31;
  int w    = threadIdx.x >> 5;
  int wy   = w >> 1, wx = w & 1;
  int row0 = blockIdx.x * 64 + wy * 32;
  int col0 = blockIdx.y * 128 + wx * 64;

  v8f acc[2][4];
  #pragma unroll
  for (int i = 0; i < 2; ++i)
    #pragma unroll
    for (int j = 0; j < 4; ++j) acc[i][j] = v8f_splat(0.f);

  for (int k0 = 0; k0 < K; k0 += 32) {
    v16h a0 = load_a_frag(A + (size_t)row0 * K + k0, K, lane);
    v16h a1 = load_a_frag(A + (size_t)(row0 + 16) * K + k0, K, lane);
    #pragma unroll
    for (int j = 0; j < 4; ++j) {
      v16h b = load_b_frag(W + (size_t)(col0 + j * 16) * K + k0, K, lane);
      acc[0][j] = __builtin_amdgcn_wmma_f32_16x16x32_f16(
          false, a0, false, b, (short)0, acc[0][j], false, false);
      acc[1][j] = __builtin_amdgcn_wmma_f32_16x16x32_f16(
          false, a1, false, b, (short)0, acc[1][j], false, false);
    }
  }

  int n = lane & 15, hh = lane >> 4;
  #pragma unroll
  for (int i = 0; i < 2; ++i)
    #pragma unroll
    for (int j = 0; j < 4; ++j)
      #pragma unroll
      for (int v = 0; v < 8; ++v) {
        size_t r = (size_t)row0 + i * 16 + hh * 8 + v;
        size_t c = (size_t)col0 + j * 16 + n;
        if (F16OUT) ((h16*)Cout)[r * N + c] = (h16)acc[i][j][v];
        else        ((float*)Cout)[r * N + c] = acc[i][j][v];
      }
}

// ---------------------------------------------------------------------------
// RoPE on Q (with 1/sqrt(DK) folded in) and K; repack [b,s,d] -> [b,h,s,dk].
// One thread per (b,h,s,pair), pair p in [0,64).
// ---------------------------------------------------------------------------
__global__ void rope_kernel(const h16* __restrict__ Qb, const h16* __restrict__ Kb,
                            const float* __restrict__ cosp, const float* __restrict__ sinp,
                            h16* __restrict__ Qh, h16* __restrict__ Kh) {
  int idx = blockIdx.x * blockDim.x + threadIdx.x;   // 2^23 total
  int p = idx & 63;
  int s = (idx >> 6) & (Sdim - 1);
  int h = (idx >> 17) & (Hn - 1);
  int b = idx >> 21;
  if (b >= Bdim) return;

  size_t src = ((size_t)b * Sdim + s) * Ddim + (size_t)h * DKd;
  float qe = (float)Qb[src + 2 * p], qo = (float)Qb[src + 2 * p + 1];
  float ke = (float)Kb[src + 2 * p], ko = (float)Kb[src + 2 * p + 1];
  float c  = cosp[s * 64 + p];
  float sn = sinp[s * 64 + p];
  const float SCALE = 0.08838834764831845f;  // 1/sqrt(128)

  size_t dst = (((size_t)b * Hn + h) * Sdim + s) * DKd;
  Qh[dst + 2 * p]     = (h16)((qe * c - qo * sn) * SCALE);
  Qh[dst + 2 * p + 1] = (h16)((qo * c + qe * sn) * SCALE);
  Kh[dst + 2 * p]     = (h16)(ke * c - ko * sn);
  Kh[dst + 2 * p + 1] = (h16)(ko * c + ke * sn);
}

// V: [b,s,d] f16 -> transposed head-major [b,h,dk,s] so P*V B-fragments load
// contiguously over the key dimension.
__global__ void vtrans_kernel(const h16* __restrict__ Vb, h16* __restrict__ Vt) {
  int idx = blockIdx.x * blockDim.x + threadIdx.x;   // 2^24 total
  int s  = idx & (Sdim - 1);
  int dk = (idx >> 11) & (DKd - 1);
  int h  = (idx >> 18) & (Hn - 1);
  int b  = idx >> 22;
  if (b >= Bdim) return;
  Vt[idx] = Vb[((size_t)b * Sdim + s) * Ddim + (size_t)h * DKd + dk];
}

// ---------------------------------------------------------------------------
// Flash attention with online softmax. 4 waves / block; each wave owns a
// 16-row tile of one (b,h). 32 keys per step: 8 WMMA for scores + 8 for P*V.
// ---------------------------------------------------------------------------
__global__ __launch_bounds__(128) void flash_attn_kernel(
    const h16* __restrict__ Qh, const h16* __restrict__ Kh,
    const h16* __restrict__ Vt, h16* __restrict__ Ob) {
  __shared__ __align__(32) h16 lds[4][16 * 32];

  int lane = threadIdx.x & 31;
  int w    = threadIdx.x >> 5;
  int bh   = blockIdx.y;                 // b*H + h
  int s0   = blockIdx.x * 64 + w * 16;   // query row tile

  const h16* Qp = Qh + ((size_t)bh * Sdim + s0) * DKd;
  const h16* Kp = Kh + (size_t)bh * Sdim * DKd;
  const h16* Vp = Vt + (size_t)bh * DKd * Sdim;

  v16h qf[4];
  #pragma unroll
  for (int kk = 0; kk < 4; ++kk) qf[kk] = load_a_frag(Qp + kk * 32, DKd, lane);

  v8f o[8];
  #pragma unroll
  for (int j = 0; j < 8; ++j) o[j] = v8f_splat(0.f);
  v8f m = v8f_splat(-1e30f);
  v8f l = v8f_splat(0.f);

  int n = lane & 15, hh = lane >> 4;
  h16* pl = lds[w];

  for (int t0 = 0; t0 <= s0 + 15; t0 += 32) {
    // ---- scores: two 16x16 tiles over 32 keys, K = DK = 4 x 32
    v8f sc0 = v8f_splat(0.f), sc1 = v8f_splat(0.f);
    #pragma unroll
    for (int kk = 0; kk < 4; ++kk) {
      v16h b0 = load_b_frag(Kp + (size_t)t0 * DKd + kk * 32, DKd, lane);
      v16h b1 = load_b_frag(Kp + (size_t)(t0 + 16) * DKd + kk * 32, DKd, lane);
      sc0 = __builtin_amdgcn_wmma_f32_16x16x32_f16(
          false, qf[kk], false, b0, (short)0, sc0, false, false);
      sc1 = __builtin_amdgcn_wmma_f32_16x16x32_f16(
          false, qf[kk], false, b1, (short)0, sc1, false, false);
    }

    // ---- causal mask + row max (reduce across 16-lane halves; lane = col)
    v8f mnew = m;
    #pragma unroll
    for (int v = 0; v < 8; ++v) {
      int srow = s0 + hh * 8 + v;
      if (t0 + n > srow)      sc0[v] = -1e30f;
      if (t0 + 16 + n > srow) sc1[v] = -1e30f;
      float mx = fmaxf(sc0[v], sc1[v]);
      #pragma unroll
      for (int msk = 1; msk < 16; msk <<= 1)
        mx = fmaxf(mx, __shfl_xor(mx, msk, 32));
      mnew[v] = fmaxf(mnew[v], mx);
    }

    // ---- exponentiate, row-sum, rescale running state
    v8f alpha;
    #pragma unroll
    for (int v = 0; v < 8; ++v) {
      alpha[v] = __expf(m[v] - mnew[v]);
      float p0 = __expf(sc0[v] - mnew[v]);
      float p1 = __expf(sc1[v] - mnew[v]);
      sc0[v] = p0; sc1[v] = p1;
      float rs = p0 + p1;
      #pragma unroll
      for (int msk = 1; msk < 16; msk <<= 1) rs += __shfl_xor(rs, msk, 32);
      l[v] = l[v] * alpha[v] + rs;
      m[v] = mnew[v];
    }
    #pragma unroll
    for (int j = 0; j < 8; ++j)
      #pragma unroll
      for (int v = 0; v < 8; ++v) o[j][v] *= alpha[v];

    // ---- P (C-layout: lane=col, reg=row) -> LDS [row][t] -> A-fragment
    #pragma unroll
    for (int v = 0; v < 8; ++v) {
      pl[(hh * 8 + v) * 32 + n]      = (h16)sc0[v];
      pl[(hh * 8 + v) * 32 + 16 + n] = (h16)sc1[v];
    }
    asm volatile("s_wait_dscnt 0" ::: "memory");
    v16h pf = load_a_frag(pl, 32, lane);

    // ---- O += P * V  (V^T stored [dk, s], contiguous over keys)
    #pragma unroll
    for (int j = 0; j < 8; ++j) {
      v16h vb = load_b_frag(Vp + (size_t)(j * 16) * Sdim + t0, Sdim, lane);
      o[j] = __builtin_amdgcn_wmma_f32_16x16x32_f16(
          false, pf, false, vb, (short)0, o[j], false, false);
    }
  }

  // ---- normalize and store f16 [b, s, h*DK + dk]
  int b = bh >> 4, h = bh & 15;
  h16* Op = Ob + ((size_t)b * Sdim + s0) * Ddim + (size_t)h * DKd;
  #pragma unroll
  for (int j = 0; j < 8; ++j)
    #pragma unroll
    for (int v = 0; v < 8; ++v) {
      float val = o[j][v] / l[v];
      Op[(size_t)(hh * 8 + v) * Ddim + j * 16 + n] = (h16)val;
    }
}

// ---------------------------------------------------------------------------
// Host-side launch. Workspace layout (192 MB, aliased stream-ordered):
//   [  0..32 MB) xh (x f16)        -> later Qh (rope'd Q, head-major)
//   [ 32..64 MB) Wq/Wk/Wv/Wo f16 (8 MB each)
//   [ 64..96 MB) Qb (proj Q)       -> later Ob (attention output)
//   [ 96..128MB) Kb (proj K)       -> later Vt (V transposed)
//   [128..160MB) Vb (proj V)
//   [160..192MB) Kh (rope'd K, head-major)
// ---------------------------------------------------------------------------
extern "C" void kernel_launch(void* const* d_in, const int* in_sizes, int n_in,
                              void* d_out, int out_size, void* d_ws, size_t ws_size,
                              hipStream_t stream) {
  const float* x    = (const float*)d_in[0];
  const float* Wq   = (const float*)d_in[1];
  const float* Wk   = (const float*)d_in[2];
  const float* Wv   = (const float*)d_in[3];
  const float* Wo   = (const float*)d_in[4];
  const float* cosp = (const float*)d_in[5];
  const float* sinp = (const float*)d_in[6];
  float* out = (float*)d_out;

  char* ws = (char*)d_ws;
  const size_t MB = 1ull << 20;
  h16* xh  = (h16*)(ws + 0 * MB);
  h16* Wqh = (h16*)(ws + 32 * MB);
  h16* Wkh = (h16*)(ws + 40 * MB);
  h16* Wvh = (h16*)(ws + 48 * MB);
  h16* Woh = (h16*)(ws + 56 * MB);
  h16* Qb  = (h16*)(ws + 64 * MB);
  h16* Kb  = (h16*)(ws + 96 * MB);
  h16* Vb  = (h16*)(ws + 128 * MB);
  h16* Kh  = (h16*)(ws + 160 * MB);
  h16* Qh  = xh;   // x no longer needed after the 3 projections
  h16* Vt  = Kb;   // Kb no longer needed after rope
  h16* Ob  = Qb;   // Qb no longer needed after rope

  // 1) fp32 -> f16
  int nx = Bdim * Sdim * Ddim;      // 16.7M
  int nw = Ddim * Ddim;             // 4.2M
  cvt_f32_to_f16<<<(nx + 255) / 256, 256, 0, stream>>>(x, xh, nx);
  cvt_f32_to_f16<<<(nw + 255) / 256, 256, 0, stream>>>(Wq, Wqh, nw);
  cvt_f32_to_f16<<<(nw + 255) / 256, 256, 0, stream>>>(Wk, Wkh, nw);
  cvt_f32_to_f16<<<(nw + 255) / 256, 256, 0, stream>>>(Wv, Wvh, nw);
  cvt_f32_to_f16<<<(nw + 255) / 256, 256, 0, stream>>>(Wo, Woh, nw);

  // 2) Q/K/V projections (WMMA GEMM, f16 out)
  dim3 gg(Mrows / 64, Ddim / 128);
  gemm_xwt_kernel<true><<<gg, 128, 0, stream>>>(xh, Wqh, Qb, Mrows, Ddim, Ddim);
  gemm_xwt_kernel<true><<<gg, 128, 0, stream>>>(xh, Wkh, Kb, Mrows, Ddim, Ddim);
  gemm_xwt_kernel<true><<<gg, 128, 0, stream>>>(xh, Wvh, Vb, Mrows, Ddim, Ddim);

  // 3) RoPE (+scale into Q) and V transpose
  int nr = Bdim * Hn * Sdim * (DKd / 2);  // 8.4M
  rope_kernel<<<(nr + 255) / 256, 256, 0, stream>>>(Qb, Kb, cosp, sinp, Qh, Kh);
  int nv = Bdim * Hn * DKd * Sdim;        // 16.7M
  vtrans_kernel<<<(nv + 255) / 256, 256, 0, stream>>>(Vb, Vt);

  // 4) causal flash attention
  dim3 ga(Sdim / 64, Bdim * Hn);
  flash_attn_kernel<<<ga, 128, 0, stream>>>(Qh, Kh, Vt, Ob);

  // 5) output projection (fp32 out)
  gemm_xwt_kernel<false><<<gg, 128, 0, stream>>>(Ob, Woh, out, Mrows, Ddim, Ddim);
}
